// attention_7438883357629
// MI455X (gfx1250) — matrix operand order repaired and checked by reading, not compiled, not run on hardware
//
#include <hip/hip_runtime.h>

// ---------------------------------------------------------------------------
// GPT-2 attention block for gfx1250 (MI455X), bf16 WMMA (16x16x32), f32 accum.
// B=2, S=2048, D=1024, H=16, hd=64.
//  - GEMMs: 4 waves/block, 32x64 tile per wave, shared 64-col B panel async-
//    staged (GLOBAL_LOAD_ASYNC_TO_LDS_B128, double buffered, ASYNCcnt).
//  - Attention: 4 waves/block share K/V 32-key tiles async-staged in LDS;
//    each wave does 32 q-rows, online softmax, 16 WMMA per key block.
// ---------------------------------------------------------------------------

typedef __attribute__((ext_vector_type(16))) __bf16 v16bf;
typedef __attribute__((ext_vector_type(8)))  float  v8f;

#define B_    2
#define S_    2048
#define D_    1024
#define H_    16
#define HD_   64
#define D3_   3072
#define ROWS_ (B_ * S_)   // 4096

union FragBF {
  v16bf v;
  uint4 q[2];
};

__device__ __forceinline__ v8f wmma_bf16(v16bf a, v16bf b, v8f c) {
  // D = A(16x32 bf16) * B(32x16 bf16) + C(16x16 f32)
  return __builtin_amdgcn_wmma_f32_16x16x32_bf16(
      false, a, false, b, (short)0, c, false, false);
}

// A fragment, 16x32 bf16, row-major source with leading dim lda.
// Per-lane: row = lane&15, contiguous K runs [8h..8h+7], [16+8h..16+8h+7].
__device__ __forceinline__ v16bf load_afrag(const __bf16* A, int row, int lda,
                                            int kbase, int half) {
  FragBF f;
  const __bf16* p = A + (size_t)row * lda + kbase;
  f.q[0] = *reinterpret_cast<const uint4*>(p + half * 8);
  f.q[1] = *reinterpret_cast<const uint4*>(p + 16 + half * 8);
  return f.v;
}

// B fragment, 32x16 bf16, source "column-major" (K contiguous per column),
// leading dim ldb. Per-lane: col = lane&15, K runs [8h..8h+7], [16+8h..16+8h+7].
__device__ __forceinline__ v16bf load_bfrag(const __bf16* Bt, int col, int ldb,
                                            int kbase, int half) {
  FragBF f;
  const __bf16* p = Bt + (size_t)col * ldb + kbase;
  f.q[0] = *reinterpret_cast<const uint4*>(p + half * 8);
  f.q[1] = *reinterpret_cast<const uint4*>(p + 16 + half * 8);
  return f.v;
}

// ---- CDNA5 async global->LDS copy (ASYNCcnt), ISA 15.18.3 opcode 98 -------
// LDS address = low 32 bits of the generic shared pointer (aperture rule).
__device__ __forceinline__ void async_ld_b128(void* lds, const void* gptr) {
  unsigned off = (unsigned)(unsigned long long)lds;
  asm volatile("global_load_async_to_lds_b128 %0, %1, off"
               :: "v"(off), "v"((unsigned long long)gptr) : "memory");
}
__device__ __forceinline__ void wait_async0() {
  asm volatile("s_wait_asynccnt 0x0" ::: "memory");
}

// ----------------------------- conversions ---------------------------------

__global__ void k_cvt_x(const float* __restrict__ x, __bf16* __restrict__ xb, int n) {
  int i = blockIdx.x * blockDim.x + threadIdx.x;
  if (i < n) xb[i] = (__bf16)x[i];
}

// Wt[f][d] = W[d][f], W is [rows=d][cols=f] f32, Wt is [cols][rows] bf16.
__global__ void k_tr_w(const float* __restrict__ W, __bf16* __restrict__ Wt,
                       int rows, int cols) {
  int i = blockIdx.x * blockDim.x + threadIdx.x;
  if (i < rows * cols) {
    int f = i / rows;
    int d = i - f * rows;
    Wt[i] = (__bf16)W[(size_t)d * cols + f];
  }
}

// ------------------------------- QKV GEMM ----------------------------------
// C[4096 x 3072] = Xb @ Wqkv + b_qkv. Block: 128 rows x 64 cols (4 waves,
// 32 rows each). B panel chunk (64 cols x 32 K = 4KB) async-staged per K-step.
__global__ void __launch_bounds__(128)
k_qkv_gemm(const __bf16* __restrict__ Xb, const __bf16* __restrict__ Wt,
           const float* __restrict__ bias, __bf16* __restrict__ q_s,
           __bf16* __restrict__ k_s, __bf16* __restrict__ v_t,
           float* __restrict__ present) {
  __shared__ __align__(16) __bf16 ldsB[2][64 * 32];  // [col][k] chunks
  const int t = threadIdx.x;
  const int wv = t >> 5, lane = t & 31;
  const int half = lane >> 4, mrow = lane & 15;
  const int CT = D3_ / 64;                  // 48 col blocks
  const int rb = blockIdx.x / CT;           // 0..31 (128-row blocks)
  const int ct = blockIdx.x - rb * CT;
  const int rt = rb * 4 + wv;               // this wave's 32-row tile

  const int arow0 = rt * 32 + mrow;
  const int arow1 = arow0 + 16;
  const __bf16* Bpanel = Wt + (size_t)(ct * 64) * D_;

  // stage helper indices: 2 chunks of 16B per thread per stage
  // chunk c (0..255): col = c>>2, off = (c&3)*8 elements
  const int c0col = t >> 2,        c0off = (t & 3) * 8;
  const int c1col = (t + 128) >> 2, c1off = ((t + 128) & 3) * 8;

  // prologue: stage K-step 0 into buffer 0
  async_ld_b128(&ldsB[0][c0col * 32 + c0off], Bpanel + (size_t)c0col * D_ + c0off);
  async_ld_b128(&ldsB[0][c1col * 32 + c1off], Bpanel + (size_t)c1col * D_ + c1off);

  v8f acc[2][4] = {};
  for (int n = 0; n < D_ / 32; ++n) {       // 32 K-steps
    const int kk = n * 32;
    wait_async0();
    __syncthreads();                        // tile n ready; tile n-1 reads done
    if (n + 1 < D_ / 32) {                  // stage next K-step (overlaps WMMA)
      const int kn = kk + 32;
      async_ld_b128(&ldsB[(n + 1) & 1][c0col * 32 + c0off],
                    Bpanel + (size_t)c0col * D_ + kn + c0off);
      async_ld_b128(&ldsB[(n + 1) & 1][c1col * 32 + c1off],
                    Bpanel + (size_t)c1col * D_ + kn + c1off);
      __builtin_prefetch(Xb + (size_t)arow0 * D_ + kn, 0, 1);
      __builtin_prefetch(Xb + (size_t)arow1 * D_ + kn, 0, 1);
    }
    const __bf16* Bt = ldsB[n & 1];
    v16bf a0 = load_afrag(Xb, arow0, D_, kk, half);
    v16bf a1 = load_afrag(Xb, arow1, D_, kk, half);
#pragma unroll
    for (int j = 0; j < 4; ++j) {
      v16bf b = load_bfrag(Bt, j * 16 + mrow, 32, 0, half);
      acc[0][j] = wmma_bf16(a0, b, acc[0][j]);
      acc[1][j] = wmma_bf16(a1, b, acc[1][j]);
    }
  }

  const int which = (ct * 64) >> 10;            // 0=Q, 1=K, 2=V (uniform)
  const int h = ((ct * 64) & (D_ - 1)) >> 6;    // head (uniform per block)
#pragma unroll
  for (int i = 0; i < 2; ++i)
#pragma unroll
    for (int j = 0; j < 4; ++j)
#pragma unroll
      for (int e = 0; e < 8; ++e) {
        const int r = rt * 32 + i * 16 + e + 8 * half;  // row in [0, 4096)
        const int f = ct * 64 + j * 16 + mrow;          // col in [0, 3072)
        const float val = acc[i][j][e] + bias[f];
        const int bb = r >> 11, s = r & (S_ - 1);
        const int dd = j * 16 + mrow;                   // 0..63 within head
        const size_t hs = ((size_t)(bb * H_ + h) * S_ + s) * HD_ + dd;
        if (which == 0) {
          q_s[hs] = (__bf16)val;
        } else if (which == 1) {
          present[(((size_t)(bb * 2 + 0) * H_ + h) * S_ + s) * HD_ + dd] = val;
          k_s[hs] = (__bf16)val;
        } else {
          present[(((size_t)(bb * 2 + 1) * H_ + h) * S_ + s) * HD_ + dd] = val;
          v_t[((size_t)(bb * H_ + h) * HD_ + dd) * S_ + s] = (__bf16)val;
        }
      }
}

// ----------------------------- flash attention ------------------------------
// Block: one (b,h) and 128 consecutive q rows (4 waves x 32 rows). K/V tiles
// (32 keys) async-staged in LDS, shared by all waves, double buffered.
__global__ void __launch_bounds__(128)
k_attn(const __bf16* __restrict__ q_s, const __bf16* __restrict__ k_s,
       const __bf16* __restrict__ v_t, __bf16* __restrict__ a_s) {
  __shared__ __align__(16) __bf16 ldsK[2][32 * 64];  // [key][dim]
  __shared__ __align__(16) __bf16 ldsV[2][64 * 32];  // [dimcol][key]
  __shared__ __align__(16) __bf16 lds_p[4][32 * 32]; // per-wave P block

  const int t = threadIdx.x;
  const int wv = t >> 5, lane = t & 31;
  const int half = lane >> 4, mrow = lane & 15;
  const int qg = blockIdx.x & 15;          // 128-row q group (S/128 = 16)
  const int bh = blockIdx.x >> 4;          // b*H + h
  const int qt = qg * 4 + wv;              // this wave's 32-row block index
  const __bf16* qp = q_s + (size_t)bh * S_ * HD_;
  const __bf16* kp = k_s + (size_t)bh * S_ * HD_;
  const __bf16* vp = v_t + (size_t)bh * HD_ * S_;

  // staging chunk maps (2 x 16B per thread per tile)
  const int kc0row = t >> 3,        kc0off = (t & 7) * 8;          // K tile
  const int kc1row = (t + 128) >> 3, kc1off = ((t + 128) & 7) * 8;
  const int vc0col = t >> 2,        vc0off = (t & 3) * 8;          // V tile
  const int vc1col = (t + 128) >> 2, vc1off = ((t + 128) & 3) * 8;

  v16bf qf[2][2];
#pragma unroll
  for (int i = 0; i < 2; ++i)
#pragma unroll
    for (int c = 0; c < 2; ++c)
      qf[i][c] = load_afrag(qp, qt * 32 + i * 16 + mrow, HD_, c * 32, half);

  float m[2][8], l[2][8];
  v8f o[2][4] = {};
#pragma unroll
  for (int i = 0; i < 2; ++i)
#pragma unroll
    for (int e = 0; e < 8; ++e) { m[i][e] = -3.0e38f; l[i][e] = 0.0f; }

  const int kbmax = qg * 4 + 3;  // last key block needed by any wave in block

  // prologue: stage key block 0
  async_ld_b128(&ldsK[0][kc0row * 64 + kc0off], kp + (size_t)kc0row * HD_ + kc0off);
  async_ld_b128(&ldsK[0][kc1row * 64 + kc1off], kp + (size_t)kc1row * HD_ + kc1off);
  async_ld_b128(&ldsV[0][vc0col * 32 + vc0off], vp + (size_t)vc0col * S_ + vc0off);
  async_ld_b128(&ldsV[0][vc1col * 32 + vc1off], vp + (size_t)vc1col * S_ + vc1off);

  for (int kb = 0; kb <= kbmax; ++kb) {
    const int kbase = kb * 32;
    wait_async0();
    __syncthreads();                 // tile kb ready; tile kb-1 reads done
    if (kb < kbmax) {                // stage next key block (overlaps WMMA)
      const int nb = (kb + 1) & 1, nbase = kbase + 32;
      async_ld_b128(&ldsK[nb][kc0row * 64 + kc0off],
                    kp + (size_t)(nbase + kc0row) * HD_ + kc0off);
      async_ld_b128(&ldsK[nb][kc1row * 64 + kc1off],
                    kp + (size_t)(nbase + kc1row) * HD_ + kc1off);
      async_ld_b128(&ldsV[nb][vc0col * 32 + vc0off],
                    vp + (size_t)vc0col * S_ + nbase + vc0off);
      async_ld_b128(&ldsV[nb][vc1col * 32 + vc1off],
                    vp + (size_t)vc1col * S_ + nbase + vc1off);
    }
    const __bf16* Kt = ldsK[kb & 1];
    const __bf16* Vt = ldsV[kb & 1];
    const bool active = (kb <= qt);  // wave-uniform: EXEC stays all-ones
    if (active) {
      // K fragments from LDS (shared by both row tiles)
      v16bf kc0t0 = load_bfrag(Kt, mrow,      64, 0,  half);
      v16bf kc1t0 = load_bfrag(Kt, mrow,      64, 32, half);
      v16bf kc0t1 = load_bfrag(Kt, 16 + mrow, 64, 0,  half);
      v16bf kc1t1 = load_bfrag(Kt, 16 + mrow, 64, 32, half);
      v8f s[2][2] = {};
#pragma unroll
      for (int i = 0; i < 2; ++i) {
        s[i][0] = wmma_bf16(qf[i][0], kc0t0, s[i][0]);
        s[i][0] = wmma_bf16(qf[i][1], kc1t0, s[i][0]);
        s[i][1] = wmma_bf16(qf[i][0], kc0t1, s[i][1]);
        s[i][1] = wmma_bf16(qf[i][1], kc1t1, s[i][1]);
      }
#pragma unroll
      for (int i = 0; i < 2; ++i)
#pragma unroll
        for (int e = 0; e < 8; ++e) {
          const int qrow = qt * 32 + i * 16 + e + 8 * half;
          const int k0 = kbase + mrow, k1 = kbase + 16 + mrow;
          float v0 = (k0 <= qrow) ? s[i][0][e] * 0.125f : -3.0e38f;
          float v1 = (k1 <= qrow) ? s[i][1][e] * 0.125f : -3.0e38f;
          float mx = fmaxf(v0, v1);
          for (int off = 1; off < 16; off <<= 1)
            mx = fmaxf(mx, __shfl_xor(mx, off, 32));
          const float mnew = fmaxf(m[i][e], mx);
          const float sc = __expf(m[i][e] - mnew);
          const float p0 = __expf(v0 - mnew);
          const float p1 = __expf(v1 - mnew);
          float ps = p0 + p1;
          for (int off = 1; off < 16; off <<= 1)
            ps += __shfl_xor(ps, off, 32);
          l[i][e] = l[i][e] * sc + ps;
          m[i][e] = mnew;
          o[i][0][e] *= sc; o[i][1][e] *= sc; o[i][2][e] *= sc; o[i][3][e] *= sc;
          const int prow = i * 16 + e + 8 * half;
          lds_p[wv][prow * 32 + mrow]      = (__bf16)p0;
          lds_p[wv][prow * 32 + 16 + mrow] = (__bf16)p1;
        }
    }
    __syncthreads();                 // uniform: P visible, K reads done
    if (active) {
      FragBF pf0, pf1;
      const __bf16* p0 = &lds_p[wv][mrow * 32];
      const __bf16* p1 = &lds_p[wv][(16 + mrow) * 32];
      pf0.q[0] = *reinterpret_cast<const uint4*>(p0 + half * 8);
      pf0.q[1] = *reinterpret_cast<const uint4*>(p0 + 16 + half * 8);
      pf1.q[0] = *reinterpret_cast<const uint4*>(p1 + half * 8);
      pf1.q[1] = *reinterpret_cast<const uint4*>(p1 + 16 + half * 8);
#pragma unroll
      for (int j = 0; j < 4; ++j) {
        v16bf bv = load_bfrag(Vt, j * 16 + mrow, 32, 0, half);
        o[0][j] = wmma_bf16(pf0.v, bv, o[0][j]);
        o[1][j] = wmma_bf16(pf1.v, bv, o[1][j]);
      }
    }
  }

  // epilogue: normalize and merge heads -> a_s bf16 [B,S,D]
  const int b = bh >> 4, h = bh & 15;
#pragma unroll
  for (int i = 0; i < 2; ++i)
#pragma unroll
    for (int e = 0; e < 8; ++e) {
      const int srow = qt * 32 + i * 16 + e + 8 * half;
      const float inv = 1.0f / l[i][e];
      const size_t base = ((size_t)b * S_ + srow) * D_ + h * 64;
      a_s[base +  0 + mrow] = (__bf16)(o[i][0][e] * inv);
      a_s[base + 16 + mrow] = (__bf16)(o[i][1][e] * inv);
      a_s[base + 32 + mrow] = (__bf16)(o[i][2][e] * inv);
      a_s[base + 48 + mrow] = (__bf16)(o[i][3][e] * inv);
    }
}

// ----------------------------- output projection ----------------------------
// out[4096 x 1024] = a_s @ W_proj + b_proj, same blocked/staged scheme.
__global__ void __launch_bounds__(128)
k_proj_gemm(const __bf16* __restrict__ Ab, const __bf16* __restrict__ Wt,
            const float* __restrict__ bias, float* __restrict__ out) {
  __shared__ __align__(16) __bf16 ldsB[2][64 * 32];
  const int t = threadIdx.x;
  const int wv = t >> 5, lane = t & 31;
  const int half = lane >> 4, mrow = lane & 15;
  const int CT = D_ / 64;                  // 16 col blocks
  const int rb = blockIdx.x / CT;
  const int ct = blockIdx.x - rb * CT;
  const int rt = rb * 4 + wv;

  const int arow0 = rt * 32 + mrow;
  const int arow1 = arow0 + 16;
  const __bf16* Bpanel = Wt + (size_t)(ct * 64) * D_;

  const int c0col = t >> 2,        c0off = (t & 3) * 8;
  const int c1col = (t + 128) >> 2, c1off = ((t + 128) & 3) * 8;

  async_ld_b128(&ldsB[0][c0col * 32 + c0off], Bpanel + (size_t)c0col * D_ + c0off);
  async_ld_b128(&ldsB[0][c1col * 32 + c1off], Bpanel + (size_t)c1col * D_ + c1off);

  v8f acc[2][4] = {};
  for (int n = 0; n < D_ / 32; ++n) {
    const int kk = n * 32;
    wait_async0();
    __syncthreads();
    if (n + 1 < D_ / 32) {
      const int kn = kk + 32;
      async_ld_b128(&ldsB[(n + 1) & 1][c0col * 32 + c0off],
                    Bpanel + (size_t)c0col * D_ + kn + c0off);
      async_ld_b128(&ldsB[(n + 1) & 1][c1col * 32 + c1off],
                    Bpanel + (size_t)c1col * D_ + kn + c1off);
      __builtin_prefetch(Ab + (size_t)arow0 * D_ + kn, 0, 1);
      __builtin_prefetch(Ab + (size_t)arow1 * D_ + kn, 0, 1);
    }
    const __bf16* Bt = ldsB[n & 1];
    v16bf a0 = load_afrag(Ab, arow0, D_, kk, half);
    v16bf a1 = load_afrag(Ab, arow1, D_, kk, half);
#pragma unroll
    for (int j = 0; j < 4; ++j) {
      v16bf b = load_bfrag(Bt, j * 16 + mrow, 32, 0, half);
      acc[0][j] = wmma_bf16(a0, b, acc[0][j]);
      acc[1][j] = wmma_bf16(a1, b, acc[1][j]);
    }
  }
#pragma unroll
  for (int i = 0; i < 2; ++i)
#pragma unroll
    for (int j = 0; j < 4; ++j)
#pragma unroll
      for (int e = 0; e < 8; ++e) {
        const int r = rt * 32 + i * 16 + e + 8 * half;
        const int f = ct * 64 + j * 16 + mrow;
        out[(size_t)r * D_ + f] = acc[i][j][e] + bias[f];
      }
}

// ------------------------------- launcher -----------------------------------

extern "C" void kernel_launch(void* const* d_in, const int* in_sizes, int n_in,
                              void* d_out, int out_size, void* d_ws, size_t ws_size,
                              hipStream_t stream) {
  const float* x      = (const float*)d_in[0];
  const float* W_qkv  = (const float*)d_in[1];
  const float* b_qkv  = (const float*)d_in[2];
  const float* W_proj = (const float*)d_in[3];
  const float* b_proj = (const float*)d_in[4];
  // d_in[5] = n_state (scalar), unused

  float* out     = (float*)d_out;
  float* present = out + (size_t)B_ * S_ * D_;  // [B,2,H,S,hd] follows out

  // workspace layout (bf16 elements); total ~50.4 MB
  __bf16* xb   = (__bf16*)d_ws;
  __bf16* wqkt = xb   + (size_t)ROWS_ * D_;        // [3072][1024]
  __bf16* wprt = wqkt + (size_t)D3_ * D_;          // [1024][1024]
  __bf16* q_s  = wprt + (size_t)D_ * D_;           // [B,H,S,hd]
  __bf16* k_s  = q_s  + (size_t)B_ * H_ * S_ * HD_;
  __bf16* v_t  = k_s  + (size_t)B_ * H_ * S_ * HD_; // [B,H,hd,S]
  __bf16* a_s  = v_t  + (size_t)B_ * H_ * S_ * HD_; // [B,S,D]

  (void)in_sizes; (void)n_in; (void)out_size; (void)ws_size;

  k_cvt_x<<<(ROWS_ * D_ + 255) / 256, 256, 0, stream>>>(x, xb, ROWS_ * D_);
  k_tr_w<<<(D3_ * D_ + 255) / 256, 256, 0, stream>>>(W_qkv, wqkt, D_, D3_);
  k_tr_w<<<(D_ * D_ + 255) / 256, 256, 0, stream>>>(W_proj, wprt, D_, D_);

  // 32 row-blocks x 48 col-blocks, 128 threads (4 waves x 32x64 tile)
  k_qkv_gemm<<<(ROWS_ / 128) * (D3_ / 64), 128, 0, stream>>>(
      xb, wqkt, b_qkv, q_s, k_s, v_t, present);

  // one block per (b, h, 128-row q group)
  k_attn<<<B_ * H_ * (S_ / 128), 128, 0, stream>>>(q_s, k_s, v_t, a_s);

  // 32 row-blocks x 16 col-blocks
  k_proj_gemm<<<(ROWS_ / 128) * (D_ / 64), 128, 0, stream>>>(
      a_s, wprt, b_proj, out);
}
